// HardThresholdingAttentionalDecoder_55722905698896
// MI455X (gfx1250) — matrix operand order repaired and checked by reading, not compiled, not run on hardware
//
#include <hip/hip_runtime.h>

typedef __attribute__((ext_vector_type(16))) _Float16 v16h;
typedef __attribute__((ext_vector_type(8)))  _Float16 h8;
typedef __attribute__((ext_vector_type(8)))  float    v8f;

#define S_LEN 2048
#define H_DIM 1024
#define NHEAD 8
#define DHEAD 128
#define BATCH 4
#define NCLS  1000

// ---------------------------------------------------------------- conversions
__global__ void cvt_f32_to_f16(const float* __restrict__ src,
                               _Float16* __restrict__ dst, int n) {
  int i = blockIdx.x * blockDim.x + threadIdx.x;
  if (i < n) dst[i] = (_Float16)src[i];
}

// dst[j*1024 + k] = (f16) src[k*1024 + j]   (1024x1024, LDS-tiled, coalesced)
__global__ void transpose_to_f16(const float* __restrict__ src,
                                 _Float16* __restrict__ dst) {
  __shared__ float tile[16][17];
  int k0 = blockIdx.x * 16, j0 = blockIdx.y * 16;
  int tx = threadIdx.x, ty = threadIdx.y;
  tile[ty][tx] = src[(size_t)(k0 + ty) * H_DIM + (j0 + tx)];
  __syncthreads();
  dst[(size_t)(j0 + ty) * H_DIM + (k0 + tx)] = (_Float16)tile[tx][ty];
}

// out[i] = bi[i] + sum_j wi[i,j]*b[j]   (folded bias for combined projection)
__global__ void bias_combine(const float* __restrict__ wi,
                             const float* __restrict__ b,
                             const float* __restrict__ bi,
                             float* __restrict__ out) {
  int i = blockIdx.x * blockDim.x + threadIdx.x;
  float acc = bi[i];
  for (int j = 0; j < H_DIM; ++j) acc += wi[(size_t)i * H_DIM + j] * b[j];
  out[i] = acc;
}

// q row s=0 per batch, full fp32 two-stage projection (4 rows total)
__global__ __launch_bounds__(256)
void q0_proj(const float* __restrict__ enc, const float* __restrict__ Wq,
             const float* __restrict__ bq, const float* __restrict__ wiq,
             const float* __restrict__ biq, float* __restrict__ q0) {
  __shared__ float xs[H_DIM];
  __shared__ float tmp[H_DIM];
  int b = blockIdx.x, t = threadIdx.x;
  const float* x = enc + (size_t)b * S_LEN * H_DIM;  // row s=0
  for (int i = t; i < H_DIM; i += 256) xs[i] = x[i];
  __syncthreads();
  for (int j = t; j < H_DIM; j += 256) {
    float a = bq[j];
    for (int k = 0; k < H_DIM; ++k) a += xs[k] * Wq[(size_t)j * H_DIM + k];
    tmp[j] = a;
  }
  __syncthreads();
  for (int i = t; i < H_DIM; i += 256) {
    float a = biq[i];
    for (int j = 0; j < H_DIM; ++j) a += tmp[j] * wiq[(size_t)i * H_DIM + j];
    q0[(size_t)b * H_DIM + i] = a;
  }
}

// ------------------------------------------------------------------ WMMA GEMM
// C[M,N] = A[M,K] (f16 row-major) @ B[N,K]^T (f16 row-major) [+ bias]
// OUT_F32 = 1 -> f32 output + bias; 0 -> f16 output (for weight combine).
// Register-blocked: each wave computes 32(M) x 64(N) with 8 f32 accumulators;
// 8 waves/block -> 64(M) x 256(N) per block. Per K-step: 12 b128 loads feed
// 8 back-to-back v_wmma_f32_16x16x32_f16.
template <int OUT_F32>
__global__ __launch_bounds__(256)
void gemm_bt_f16(const _Float16* __restrict__ A, const _Float16* __restrict__ Bm,
                 const float* __restrict__ bias, void* __restrict__ out,
                 int M, int N, int K) {
  int wave = threadIdx.x >> 5;
  int lane = threadIdx.x & 31;
  int r16 = lane & 15;
  int hi  = lane >> 4;  // 0 or 1
  int m0 = blockIdx.x * 64 + (wave & 1) * 32;   // wave: 32 rows
  int n0 = blockIdx.y * 256 + (wave >> 1) * 64; // wave: 64 cols

  // per-lane row pointers (A: two 16-row fragments; B: four 16-col fragments)
  const _Float16* ar0 = A  + (size_t)(m0 + r16) * K;
  const _Float16* ar1 = A  + (size_t)(m0 + 16 + r16) * K;
  const _Float16* br0 = Bm + (size_t)(n0 + r16) * K;
  const _Float16* br1 = Bm + (size_t)(n0 + 16 + r16) * K;
  const _Float16* br2 = Bm + (size_t)(n0 + 32 + r16) * K;
  const _Float16* br3 = Bm + (size_t)(n0 + 48 + r16) * K;

  v8f c00 = {}, c01 = {}, c02 = {}, c03 = {};
  v8f c10 = {}, c11 = {}, c12 = {}, c13 = {};

  const int ao = hi * 16;  // A frag K sub-offset for this lane half
  const int bo = hi * 16;  // B frag K sub-offset for this lane half

#pragma unroll 2
  for (int kk = 0; kk < K; kk += 32) {
    // prefetch 4 K-steps ahead (speculative; OOB silently dropped)
    __builtin_prefetch(ar0 + kk + 128, 0, 1);
    __builtin_prefetch(ar1 + kk + 128, 0, 1);
    __builtin_prefetch(br0 + kk + 128, 0, 1);
    __builtin_prefetch(br2 + kk + 128, 0, 1);

    h8 a0l = *(const h8*)(ar0 + kk + ao);
    h8 a0h = *(const h8*)(ar0 + kk + ao + 8);
    h8 a1l = *(const h8*)(ar1 + kk + ao);
    h8 a1h = *(const h8*)(ar1 + kk + ao + 8);
    h8 b0l = *(const h8*)(br0 + kk + bo);
    h8 b0h = *(const h8*)(br0 + kk + bo + 8);
    h8 b1l = *(const h8*)(br1 + kk + bo);
    h8 b1h = *(const h8*)(br1 + kk + bo + 8);
    h8 b2l = *(const h8*)(br2 + kk + bo);
    h8 b2h = *(const h8*)(br2 + kk + bo + 8);
    h8 b3l = *(const h8*)(br3 + kk + bo);
    h8 b3h = *(const h8*)(br3 + kk + bo + 8);

    v16h a0 = __builtin_shufflevector(a0l, a0h, 0,1,2,3,4,5,6,7,8,9,10,11,12,13,14,15);
    v16h a1 = __builtin_shufflevector(a1l, a1h, 0,1,2,3,4,5,6,7,8,9,10,11,12,13,14,15);
    v16h b0 = __builtin_shufflevector(b0l, b0h, 0,1,2,3,4,5,6,7,8,9,10,11,12,13,14,15);
    v16h b1 = __builtin_shufflevector(b1l, b1h, 0,1,2,3,4,5,6,7,8,9,10,11,12,13,14,15);
    v16h b2 = __builtin_shufflevector(b2l, b2h, 0,1,2,3,4,5,6,7,8,9,10,11,12,13,14,15);
    v16h b3 = __builtin_shufflevector(b3l, b3h, 0,1,2,3,4,5,6,7,8,9,10,11,12,13,14,15);

    c00 = __builtin_amdgcn_wmma_f32_16x16x32_f16(false, a0, false, b0, (short)0, c00, false, false);
    c01 = __builtin_amdgcn_wmma_f32_16x16x32_f16(false, a0, false, b1, (short)0, c01, false, false);
    c02 = __builtin_amdgcn_wmma_f32_16x16x32_f16(false, a0, false, b2, (short)0, c02, false, false);
    c03 = __builtin_amdgcn_wmma_f32_16x16x32_f16(false, a0, false, b3, (short)0, c03, false, false);
    c10 = __builtin_amdgcn_wmma_f32_16x16x32_f16(false, a1, false, b0, (short)0, c10, false, false);
    c11 = __builtin_amdgcn_wmma_f32_16x16x32_f16(false, a1, false, b1, (short)0, c11, false, false);
    c12 = __builtin_amdgcn_wmma_f32_16x16x32_f16(false, a1, false, b2, (short)0, c12, false, false);
    c13 = __builtin_amdgcn_wmma_f32_16x16x32_f16(false, a1, false, b3, (short)0, c13, false, false);
  }

  v8f acc[2][4] = {{c00, c01, c02, c03}, {c10, c11, c12, c13}};
#pragma unroll
  for (int i = 0; i < 2; ++i) {
#pragma unroll
    for (int j = 0; j < 4; ++j) {
      int n = n0 + j * 16 + r16;
      int mb = m0 + i * 16 + hi * 8;
      if (OUT_F32) {
        float bv = bias ? bias[n] : 0.f;
        float* o = (float*)out;
#pragma unroll
        for (int r = 0; r < 8; ++r)
          o[(size_t)(mb + r) * N + n] = acc[i][j][r] + bv;
      } else {
        _Float16* o = (_Float16*)out;
#pragma unroll
        for (int r = 0; r < 8; ++r)
          o[(size_t)(mb + r) * N + n] = (_Float16)acc[i][j][r];
      }
    }
  }
}

// ---------------------------------------------- attention for query row 0 only
// One block per (b,h). Scores/softmax/threshold in fp32; writes cls weights.
__global__ __launch_bounds__(256)
void attn_q0(const float* __restrict__ q0, const float* __restrict__ Kb,
             const float* __restrict__ Vb, const int* __restrict__ mask,
             float* __restrict__ attn0, float* __restrict__ wout) {
  __shared__ float qs[DHEAD];
  __shared__ float sc[S_LEN];
  __shared__ float red[256];
  int bh = blockIdx.x, b = bh >> 3, h = bh & 7;
  int t = threadIdx.x;
  for (int i = t; i < DHEAD; i += 256) qs[i] = q0[(size_t)b * H_DIM + h * DHEAD + i];
  __syncthreads();
  const float scale = 0.0883883476483184f;  // 1/sqrt(128)
  for (int k = t; k < S_LEN; k += 256) {
    const float* kr = Kb + (size_t)(b * S_LEN + k) * H_DIM + h * DHEAD;
    float s = 0.f;
    for (int i = 0; i < DHEAD; ++i) s += qs[i] * kr[i];
    s *= scale;
    if (mask[b * S_LEN + k] == 0) s = -1e9f;
    sc[k] = s;
  }
  __syncthreads();
  // row max
  float v = -3.4e38f;
  for (int k = t; k < S_LEN; k += 256) v = fmaxf(v, sc[k]);
  red[t] = v; __syncthreads();
  for (int s = 128; s > 0; s >>= 1) { if (t < s) red[t] = fmaxf(red[t], red[t + s]); __syncthreads(); }
  float m = red[0]; __syncthreads();
  // Z = sum exp
  v = 0.f;
  for (int k = t; k < S_LEN; k += 256) { float e = expf(sc[k] - m); sc[k] = e; v += e; }
  red[t] = v; __syncthreads();
  for (int s = 128; s > 0; s >>= 1) { if (t < s) red[t] += red[t + s]; __syncthreads(); }
  float Z = red[0]; __syncthreads();
  // survivors: w = e/Z >= 0.001  <=>  e >= 0.001*Z ; T = sum of surviving e
  float thr = 0.001f * Z;
  v = 0.f;
  for (int k = t; k < S_LEN; k += 256) { float e = sc[k]; e = (e >= thr) ? e : 0.f; sc[k] = e; v += e; }
  red[t] = v; __syncthreads();
  for (int s = 128; s > 0; s >>= 1) { if (t < s) red[t] += red[t + s]; __syncthreads(); }
  float T = red[0]; __syncthreads();
  // w' = e*keep / (T + 1e-9*Z)  ==  (e/Z*keep) / (T/Z + 1e-9)
  float inv = 1.f / (T + 1e-9f * Z);
  for (int k = t; k < S_LEN; k += 256) {
    float ww = sc[k] * inv;
    sc[k] = ww;
    wout[(size_t)bh * S_LEN + k] = ww;  // cls_attn_weights[b,h,k]
  }
  __syncthreads();
  // out[i] = sum_k w'_k * V[k,i]
  int i = t & 127, half = t >> 7;
  float acc = 0.f;
  for (int k = half * 1024; k < half * 1024 + 1024; ++k)
    acc += sc[k] * Vb[(size_t)(b * S_LEN + k) * H_DIM + h * DHEAD + i];
  red[t] = acc; __syncthreads();
  if (t < 128) attn0[(size_t)b * H_DIM + h * DHEAD + i] = red[t] + red[t + 128];
}

// --------------------------- out_proj + residual + LayerNorm + FC (4 rows)
__global__ __launch_bounds__(256)
void final_head(const float* __restrict__ attn0, const float* __restrict__ enc,
                const float* __restrict__ Wo, const float* __restrict__ bo,
                const float* __restrict__ g, const float* __restrict__ be,
                const float* __restrict__ fw, const float* __restrict__ fb,
                float* __restrict__ logits) {
  __shared__ float ai[H_DIM];
  __shared__ float xn[H_DIM];
  __shared__ float red[256];
  int b = blockIdx.x, t = threadIdx.x;
  for (int i = t; i < H_DIM; i += 256) ai[i] = attn0[(size_t)b * H_DIM + i];
  __syncthreads();
  for (int n = t; n < H_DIM; n += 256) {
    float a = bo[n];
    for (int j = 0; j < H_DIM; ++j) a += ai[j] * Wo[(size_t)n * H_DIM + j];
    xn[n] = a + enc[(size_t)b * S_LEN * H_DIM + n];  // residual, row s=0
  }
  __syncthreads();
  float s = 0.f;
  for (int i = t; i < H_DIM; i += 256) s += xn[i];
  red[t] = s; __syncthreads();
  for (int k = 128; k > 0; k >>= 1) { if (t < k) red[t] += red[t + k]; __syncthreads(); }
  float mu = red[0] / H_DIM; __syncthreads();
  s = 0.f;
  for (int i = t; i < H_DIM; i += 256) { float d = xn[i] - mu; s += d * d; }
  red[t] = s; __syncthreads();
  for (int k = 128; k > 0; k >>= 1) { if (t < k) red[t] += red[t + k]; __syncthreads(); }
  float rstd = rsqrtf(red[0] / H_DIM + 1e-5f); __syncthreads();
  for (int i = t; i < H_DIM; i += 256) xn[i] = (xn[i] - mu) * rstd * g[i] + be[i];
  __syncthreads();
  for (int c = t; c < NCLS; c += 256) {
    float a = fb[c];
    for (int j = 0; j < H_DIM; ++j) a += xn[j] * fw[(size_t)c * H_DIM + j];
    logits[(size_t)b * NCLS + c] = a;
  }
}

// ------------------------------------------------------------------- launcher
extern "C" void kernel_launch(void* const* d_in, const int* in_sizes, int n_in,
                              void* d_out, int out_size, void* d_ws, size_t ws_size,
                              hipStream_t stream) {
  const float* enc  = (const float*)d_in[0];
  const int*   mask = (const int*)d_in[1];
  const float* Wq   = (const float*)d_in[2];
  const float* bq   = (const float*)d_in[3];
  const float* Wk   = (const float*)d_in[4];
  const float* bk   = (const float*)d_in[5];
  const float* Wv   = (const float*)d_in[6];
  const float* bv   = (const float*)d_in[7];
  const float* ipw  = (const float*)d_in[8];
  const float* ipb  = (const float*)d_in[9];
  const float* Wo   = (const float*)d_in[10];
  const float* bo   = (const float*)d_in[11];
  const float* lng  = (const float*)d_in[12];
  const float* lnb  = (const float*)d_in[13];
  const float* fw   = (const float*)d_in[14];
  const float* fb   = (const float*)d_in[15];
  float* out = (float*)d_out;  // [0,4000): logits ; [4000,69536): cls attn (B,8,S)

  char* wsp = (char*)d_ws;
  size_t off = 0;
  auto alloc = [&](size_t bytes) {
    void* p = wsp + off;
    off = (off + bytes + 255) & ~(size_t)255;
    return p;
  };
  _Float16* x16   = (_Float16*)alloc((size_t)BATCH * S_LEN * H_DIM * 2);
  _Float16* wik16 = (_Float16*)alloc((size_t)H_DIM * H_DIM * 2);
  _Float16* wiv16 = (_Float16*)alloc((size_t)H_DIM * H_DIM * 2);
  _Float16* WkT16 = (_Float16*)alloc((size_t)H_DIM * H_DIM * 2);
  _Float16* WvT16 = (_Float16*)alloc((size_t)H_DIM * H_DIM * 2);
  _Float16* Wk2   = (_Float16*)alloc((size_t)H_DIM * H_DIM * 2);
  _Float16* Wv2   = (_Float16*)alloc((size_t)H_DIM * H_DIM * 2);
  float*    b2k   = (float*)alloc(H_DIM * 4);
  float*    b2v   = (float*)alloc(H_DIM * 4);
  float*    q0    = (float*)alloc((size_t)BATCH * H_DIM * 4);
  float*    Kbuf  = (float*)alloc((size_t)BATCH * S_LEN * H_DIM * 4);
  float*    Vbuf  = (float*)alloc((size_t)BATCH * S_LEN * H_DIM * 4);
  float*    attn0 = (float*)alloc((size_t)BATCH * H_DIM * 4);

  const int nX = BATCH * S_LEN * H_DIM;   // 8388608
  const int nW = H_DIM * H_DIM;           // 1048576

  // precision prep (f16 operands for WMMA)
  cvt_f32_to_f16<<<nX / 256, 256, 0, stream>>>(enc, x16, nX);
  cvt_f32_to_f16<<<nW / 256, 256, 0, stream>>>(ipw + (size_t)1 * nW, wik16, nW);
  cvt_f32_to_f16<<<nW / 256, 256, 0, stream>>>(ipw + (size_t)2 * nW, wiv16, nW);
  transpose_to_f16<<<dim3(64, 64), dim3(16, 16), 0, stream>>>(Wk, WkT16);
  transpose_to_f16<<<dim3(64, 64), dim3(16, 16), 0, stream>>>(Wv, WvT16);

  // folded biases: b2 = wi@b + bi
  bias_combine<<<4, 256, 0, stream>>>(ipw + (size_t)1 * nW, bk, ipb + H_DIM, b2k);
  bias_combine<<<4, 256, 0, stream>>>(ipw + (size_t)2 * nW, bv, ipb + 2 * H_DIM, b2v);

  // q row 0 per batch (fp32 exact)
  q0_proj<<<4, 256, 0, stream>>>(enc, Wq, bq, ipw, ipb, q0);

  // combined weights: W2 = wi @ W  (A=wi row-major, B=W^T -> A@B^T) [WMMA]
  gemm_bt_f16<0><<<dim3(16, 4), 256, 0, stream>>>(wik16, WkT16, nullptr, Wk2,
                                                  H_DIM, H_DIM, H_DIM);
  gemm_bt_f16<0><<<dim3(16, 4), 256, 0, stream>>>(wiv16, WvT16, nullptr, Wv2,
                                                  H_DIM, H_DIM, H_DIM);

  // main projections: K/V = x @ W2^T + b2  (8192x1024x1024 each) [WMMA]
  gemm_bt_f16<1><<<dim3(128, 4), 256, 0, stream>>>(x16, Wk2, b2k, Kbuf,
                                                   BATCH * S_LEN, H_DIM, H_DIM);
  gemm_bt_f16<1><<<dim3(128, 4), 256, 0, stream>>>(x16, Wv2, b2v, Vbuf,
                                                   BATCH * S_LEN, H_DIM, H_DIM);

  // attention for query 0; writes cls weights into d_out[4000..]
  attn_q0<<<BATCH * NHEAD, 256, 0, stream>>>(q0, Kbuf, Vbuf, mask, attn0,
                                             out + BATCH * NCLS);

  // out_proj + residual + LN + FC -> logits
  final_head<<<BATCH, 256, 0, stream>>>(attn0, enc, Wo, bo, lng, lnb, fw, fb, out);

  (void)in_sizes; (void)n_in; (void)out_size; (void)ws_size;
}
